// MPNNLatencyPredictor_42210938585393
// MI455X (gfx1250) — compile-verified, matrix operand order checked by reading, not run on hardware
//
#include <hip/hip_runtime.h>
#include <hip/hip_bf16.h>
#include <math.h>

typedef __attribute__((ext_vector_type(16))) _Float16 v16h;
typedef __attribute__((ext_vector_type(8)))  float    v8f;

#define NN 32768
#define EE 32768
#define NDIM 12
#define EDIM 5
#define GDIM 11
#define HH 64
#define KB_MSG 130   // K = 4160 = 65*64 (last 64-row slice carries be2)

// ---------------- WMMA fragment helpers (CDNA5 16x16x32 f16 layout) ----------------
// A (16x32 f16): lane<16 -> M=lane, halves q hold K = {0..7,16..23}; lane>=16 -> +8.
__device__ __forceinline__ int k_of_half(int lane, int q) {
  return (((q & 8) ? (16 + (q & 7)) : q) + ((lane & 16) ? 8 : 0));
}

__device__ __forceinline__ v16h make_a_frag(const float* rows, int lane, int kbase, int stride) {
  int m = lane & 15;
  v16h a;
#pragma unroll
  for (int q = 0; q < 16; ++q) {
    int k = kbase + k_of_half(lane, q);
    a[q] = (_Float16)rows[m * stride + k];
  }
  return a;
}

// B fragments are pre-packed in exactly per-lane register order:
// half index = ((kb*NB + nb)*32 + lane)*16 + q  -> one 32-byte contiguous load per lane.
__device__ __forceinline__ v16h load_b_frag(const _Float16* packed, int NB, int kb, int nb, int lane) {
  return *(const v16h*)(packed + ((size_t)(kb * NB + nb) * 32 + lane) * 16);
}

__device__ __forceinline__ v8f wmma_f16(v16h a, v16h b, v8f c) {
  return __builtin_amdgcn_wmma_f32_16x16x32_f16(false, a, false, b, (short)0, c, false, false);
}

__device__ __forceinline__ v16h splat16(_Float16 x) {
  v16h v;
#pragma unroll
  for (int q = 0; q < 16; ++q) v[q] = x;
  return v;
}

// ---------------- weight packing (runs once per launch, deterministic) ----------------
// arr0 Bmsg : logical (4160 x 64). row k: kg=k>>6, hh=k&63.
//            kg<64 -> We2[(hh*64+o)*64+kg]; kg==64 -> be2[hh*64+o]  (bias fold)
// arr1 WB1  : (64 x 256) = [root | Whh^T]   (h @ root, h @ Whh^T fused)
// arr2 WB2  : (64 x 192) = Wih^T            (m @ Wih^T)
// arr3 WD1  : (160 x 64) = Wd1^T zero-padded K 133->160
// arr4 WD2  : (64 x 32)  = Wd2^T
__global__ void pack_weights(const float* We2, const float* be2,
                             const float* root, const float* Whh, const float* Wih,
                             const float* Wd1, const float* Wd2,
                             _Float16* Bmsg, _Float16* WB1, _Float16* WB2,
                             _Float16* WD1, _Float16* WD2) {
  int idx = blockIdx.x * blockDim.x + threadIdx.x;
  const int S0 = KB_MSG * 4 * 512;  // 266240
  const int S1 = 2 * 16 * 512;      // 16384
  const int S2 = 2 * 12 * 512;      // 12288
  const int S3 = 5 * 4 * 512;       // 10240
  const int S4 = 2 * 2 * 512;       // 2048
  int arr, NB, local;
  _Float16* dst;
  if (idx < S0)                { arr = 0; NB = 4;  local = idx;              dst = Bmsg; }
  else if (idx < S0+S1)        { arr = 1; NB = 16; local = idx-S0;           dst = WB1; }
  else if (idx < S0+S1+S2)     { arr = 2; NB = 12; local = idx-S0-S1;        dst = WB2; }
  else if (idx < S0+S1+S2+S3)  { arr = 3; NB = 4;  local = idx-S0-S1-S2;     dst = WD1; }
  else if (idx < S0+S1+S2+S3+S4){arr = 4; NB = 2;  local = idx-S0-S1-S2-S3;  dst = WD2; }
  else return;
  int q    = local & 15;
  int lane = (local >> 4) & 31;
  int tile = local >> 9;
  int nb = tile % NB;
  int kb = tile / NB;
  int k = kb * 32 + k_of_half(lane, q);
  int n = nb * 16 + (lane & 15);
  float v = 0.f;
  if (arr == 0) {
    int kg = k >> 6, hh = k & 63;
    v = (kg < 64) ? We2[(hh * 64 + n) * 64 + kg] : be2[hh * 64 + n];
  } else if (arr == 1) {
    v = (n < 64) ? root[k * 64 + n] : Whh[(n - 64) * 64 + k];
  } else if (arr == 2) {
    v = Wih[n * 64 + k];
  } else if (arr == 3) {
    v = (k < 133) ? Wd1[n * 133 + k] : 0.f;
  } else {
    v = Wd2[n * 64 + k];
  }
  dst[local] = (_Float16)v;
}

// ---------------- tiny element-wise kernels ----------------
__global__ void node_proj(const float* x, const float* u, const float* Wp, const float* bp, float* h) {
  int tid = blockIdx.x * blockDim.x + threadIdx.x;
  if (tid >= NN * HH) return;
  int n = tid >> 6, o = tid & 63;
  float acc = bp[o];
  const float* w = Wp + o * (NDIM + GDIM);
#pragma unroll
  for (int j = 0; j < NDIM; ++j) acc += x[n * NDIM + j] * w[j];
#pragma unroll
  for (int j = 0; j < GDIM; ++j) acc += u[j] * w[NDIM + j];
  h[tid] = tanhf(acc);
}

__global__ void edge_gate(const float* ea, const float* We1, const float* be1, float* g) {
  int tid = blockIdx.x * blockDim.x + threadIdx.x;
  if (tid >= EE * HH) return;
  int e = tid >> 6, o = tid & 63;
  float acc = be1[o];
#pragma unroll
  for (int j = 0; j < EDIM; ++j) acc += ea[e * EDIM + j] * We1[o * EDIM + j];
  g[tid] = fmaxf(acc, 0.f);
}

__global__ void count_dst(const int* ei, float* counts) {
  int e = blockIdx.x * blockDim.x + threadIdx.x;
  if (e >= EE) return;
  atomicAdd(&counts[ei[EE + e]], 1.0f);
}

__global__ void scatter_msg(const float* msg, const int* ei, float* agg) {
  int tid = blockIdx.x * blockDim.x + threadIdx.x;
  if (tid >= EE * HH) return;
  int e = tid >> 6, o = tid & 63;
  atomicAdd(&agg[ei[EE + e] * HH + o], msg[tid]);
}

// ---------------- msg = rowwise-outer(g, h[src]) @ Bmsg  (K = 4160, WMMA) ----------------
// One wave handles a 32-edge x 64-col tile (two 16-row groups sharing B fragments).
// hs rows live in f16 fragment-order registers; per K-block A = hf * splat(g) (pk_mul_f16).
// B fragments are double-buffered: next K-block's loads issue before current WMMAs.
__global__ void __launch_bounds__(32) msg_gemm(const float* h, const float* g, const int* ei,
                                               const _Float16* Bmsg, float* msg) {
  __shared__ float hs_s[32 * 64];
  __shared__ float gs_s[32 * 64];
  int lane = threadIdx.x;
  int base = blockIdx.x * 32;
#pragma unroll
  for (int i = 0; i < 64; ++i) {
    int idx = lane + i * 32;
    int row = idx >> 6, col = idx & 63;
    int e = base + row;
    hs_s[idx] = h[ei[e] * HH + col];
    gs_s[idx] = g[e * HH + col];
  }
  __syncthreads();
  int m = lane & 15;
  // hoist hs rows (frag order, f16) once: rows m (group A) and m+16 (group B),
  // for both 32-wide K halves of the 64-wide h vector.
  v16h hfa0, hfa1, hfb0, hfb1;
#pragma unroll
  for (int q = 0; q < 16; ++q) {
    int kl = k_of_half(lane, q);
    hfa0[q] = (_Float16)hs_s[m * 64 + kl];
    hfa1[q] = (_Float16)hs_s[m * 64 + 32 + kl];
    hfb0[q] = (_Float16)hs_s[(m + 16) * 64 + kl];
    hfb1[q] = (_Float16)hs_s[(m + 16) * 64 + 32 + kl];
  }
  v8f c0 = {}, c1 = {}, c2 = {}, c3 = {};   // rows 0..15
  v8f c4 = {}, c5 = {}, c6 = {}, c7 = {};   // rows 16..31
  // prime B double-buffer with kb = 0
  v16h b0 = load_b_frag(Bmsg, 4, 0, 0, lane);
  v16h b1 = load_b_frag(Bmsg, 4, 0, 1, lane);
  v16h b2 = load_b_frag(Bmsg, 4, 0, 2, lane);
  v16h b3 = load_b_frag(Bmsg, 4, 0, 3, lane);
  for (int kg = 0; kg < 65; ++kg) {         // kb = 2*kg, 2*kg+1
    float ga = (kg < 64) ? gs_s[m * 64 + kg] : 1.0f;         // last slice = be2 fold
    float gb = (kg < 64) ? gs_s[(m + 16) * 64 + kg] : 1.0f;
    v16h gva = splat16((_Float16)ga);
    v16h gvb = splat16((_Float16)gb);
    {                                       // kb = 2*kg (uses b*, prefetch 2*kg+1)
      int nkb = 2 * kg + 1;
      v16h p0 = load_b_frag(Bmsg, 4, nkb, 0, lane);
      v16h p1 = load_b_frag(Bmsg, 4, nkb, 1, lane);
      v16h p2 = load_b_frag(Bmsg, 4, nkb, 2, lane);
      v16h p3 = load_b_frag(Bmsg, 4, nkb, 3, lane);
      v16h aA = hfa0 * gva;
      v16h aB = hfb0 * gvb;
      c0 = wmma_f16(aA, b0, c0);  c1 = wmma_f16(aA, b1, c1);
      c2 = wmma_f16(aA, b2, c2);  c3 = wmma_f16(aA, b3, c3);
      c4 = wmma_f16(aB, b0, c4);  c5 = wmma_f16(aB, b1, c5);
      c6 = wmma_f16(aB, b2, c6);  c7 = wmma_f16(aB, b3, c7);
      b0 = p0; b1 = p1; b2 = p2; b3 = p3;
    }
    {                                       // kb = 2*kg+1 (uses b*, prefetch 2*kg+2)
      int nkb = (kg < 64) ? (2 * kg + 2) : (2 * kg + 1);   // last iter: harmless reload
      v16h p0 = load_b_frag(Bmsg, 4, nkb, 0, lane);
      v16h p1 = load_b_frag(Bmsg, 4, nkb, 1, lane);
      v16h p2 = load_b_frag(Bmsg, 4, nkb, 2, lane);
      v16h p3 = load_b_frag(Bmsg, 4, nkb, 3, lane);
      v16h aA = hfa1 * gva;
      v16h aB = hfb1 * gvb;
      c0 = wmma_f16(aA, b0, c0);  c1 = wmma_f16(aA, b1, c1);
      c2 = wmma_f16(aA, b2, c2);  c3 = wmma_f16(aA, b3, c3);
      c4 = wmma_f16(aB, b0, c4);  c5 = wmma_f16(aB, b1, c5);
      c6 = wmma_f16(aB, b2, c6);  c7 = wmma_f16(aB, b3, c7);
      b0 = p0; b1 = p1; b2 = p2; b3 = p3;
    }
  }
  int mro  = (lane & 16) ? 8 : 0;
  int ncol = lane & 15;
#pragma unroll
  for (int r = 0; r < 8; ++r) {
    int rowA = base + mro + r;
    msg[rowA * HH +  0 + ncol] = c0[r];
    msg[rowA * HH + 16 + ncol] = c1[r];
    msg[rowA * HH + 32 + ncol] = c2[r];
    msg[rowA * HH + 48 + ncol] = c3[r];
    int rowB = base + 16 + mro + r;
    msg[rowB * HH +  0 + ncol] = c4[r];
    msg[rowB * HH + 16 + ncol] = c5[r];
    msg[rowB * HH + 32 + ncol] = c6[r];
    msg[rowB * HH + 48 + ncol] = c7[r];
  }
}

// ---------------- fused NNConv-combine + GRU cell for 16 nodes per wave ----------------
__global__ void __launch_bounds__(32) node_update(float* h, const float* agg, const float* counts,
                                                  const float* conv_b, const float* bih, const float* bhh,
                                                  const _Float16* WB1, const _Float16* WB2) {
  __shared__ float hv_s[16 * 64];
  __shared__ float m_s [16 * 64];
  __shared__ float gh_s[16 * 192];
  __shared__ float gi_s[16 * 192];
  int lane = threadIdx.x;
  int base = blockIdx.x * 16;
#pragma unroll
  for (int i = 0; i < 32; ++i) {
    int idx = lane + i * 32;
    hv_s[idx] = h[base * HH + idx];
  }
  __syncthreads();
  v16h a0 = make_a_frag(hv_s, lane, 0, 64);
  v16h a1 = make_a_frag(hv_s, lane, 32, 64);
  int mro  = (lane & 16) ? 8 : 0;
  int ncol = lane & 15;
#pragma unroll
  for (int nb = 0; nb < 16; ++nb) {   // cols 0..63: h@root ; cols 64..255: h@Whh^T
    v8f c = {};
    c = wmma_f16(a0, load_b_frag(WB1, 16, 0, nb, lane), c);
    c = wmma_f16(a1, load_b_frag(WB1, 16, 1, nb, lane), c);
    int j = nb * 16 + ncol;
#pragma unroll
    for (int r = 0; r < 8; ++r) {
      int row = mro + r;
      if (j < 64) {
        int n = base + row;
        float cnt  = fmaxf(counts[n], 1.0f);
        float aval = agg[n * HH + j] / cnt;
        m_s[row * 64 + j] = fmaxf(aval + c[r] + conv_b[j], 0.0f);
      } else {
        gh_s[row * 192 + (j - 64)] = c[r] + bhh[j - 64];
      }
    }
  }
  __syncthreads();
  a0 = make_a_frag(m_s, lane, 0, 64);
  a1 = make_a_frag(m_s, lane, 32, 64);
#pragma unroll
  for (int nb = 0; nb < 12; ++nb) {   // gi = m @ Wih^T
    v8f c = {};
    c = wmma_f16(a0, load_b_frag(WB2, 12, 0, nb, lane), c);
    c = wmma_f16(a1, load_b_frag(WB2, 12, 1, nb, lane), c);
    int j = nb * 16 + ncol;
#pragma unroll
    for (int r = 0; r < 8; ++r) gi_s[(mro + r) * 192 + j] = c[r] + bih[j];
  }
  __syncthreads();
#pragma unroll
  for (int i = 0; i < 32; ++i) {      // GRU gates (torch order r,z,n)
    int idx = lane + i * 32;
    int row = idx >> 6, o = idx & 63;
    float rg = 1.f / (1.f + expf(-(gi_s[row*192 +       o] + gh_s[row*192 +       o])));
    float zg = 1.f / (1.f + expf(-(gi_s[row*192 +  64 + o] + gh_s[row*192 +  64 + o])));
    float ng = tanhf(gi_s[row*192 + 128 + o] + rg * gh_s[row*192 + 128 + o]);
    h[base * HH + idx] = (1.f - zg) * ng + zg * hv_s[idx];
  }
}

// ---------------- per-edge decoder MLP (WMMA for layers 1-2, VALU tail) ----------------
__global__ void __launch_bounds__(32) edge_decode(const float* h, const float* ea, const int* ei,
                                                  const _Float16* WD1, const _Float16* WD2,
                                                  const float* bd1, const float* bd2,
                                                  const float* Wd3, const float* bd3, float* out) {
  __shared__ float ein_s[16 * 160];
  __shared__ float d1_s [16 * 64];
  __shared__ float d2_s [16 * 32];
  int lane = threadIdx.x;
  int base = blockIdx.x * 16;
#pragma unroll
  for (int i = 0; i < 80; ++i) {
    int idx = lane + i * 32;
    int row = idx / 160, col = idx % 160;
    int e = base + row;
    float v;
    if      (col <  64) v = h[ei[e]      * HH + col];
    else if (col < 128) v = h[ei[EE + e] * HH + (col - 64)];
    else if (col < 133) v = ea[e * EDIM + (col - 128)];
    else                v = 0.f;
    ein_s[idx] = v;
  }
  __syncthreads();
  v16h af[5];
#pragma unroll
  for (int kb = 0; kb < 5; ++kb) af[kb] = make_a_frag(ein_s, lane, kb * 32, 160);
  int mro  = (lane & 16) ? 8 : 0;
  int ncol = lane & 15;
#pragma unroll
  for (int nb = 0; nb < 4; ++nb) {
    v8f c = {};
#pragma unroll
    for (int kb = 0; kb < 5; ++kb) c = wmma_f16(af[kb], load_b_frag(WD1, 4, kb, nb, lane), c);
    int j = nb * 16 + ncol;
#pragma unroll
    for (int r = 0; r < 8; ++r) d1_s[(mro + r) * 64 + j] = fmaxf(c[r] + bd1[j], 0.f);
  }
  __syncthreads();
  v16h a0 = make_a_frag(d1_s, lane, 0, 64);
  v16h a1 = make_a_frag(d1_s, lane, 32, 64);
#pragma unroll
  for (int nb = 0; nb < 2; ++nb) {
    v8f c = {};
    c = wmma_f16(a0, load_b_frag(WD2, 2, 0, nb, lane), c);
    c = wmma_f16(a1, load_b_frag(WD2, 2, 1, nb, lane), c);
    int j = nb * 16 + ncol;
#pragma unroll
    for (int r = 0; r < 8; ++r) d2_s[(mro + r) * 32 + j] = fmaxf(c[r] + bd2[j], 0.f);
  }
  __syncthreads();
#pragma unroll
  for (int p = 0; p < 2; ++p) {       // final 32->4 layer: 64 (edge,t) pairs over 32 lanes
    int idx = lane * 2 + p;
    int row = idx >> 2, t = idx & 3;
    float acc = bd3[t];
#pragma unroll
    for (int k = 0; k < 32; ++k) acc += d2_s[row * 32 + k] * Wd3[t * 32 + k];
    out[(base + row) * 4 + t] = acc;
  }
}

// ---------------- launch ----------------
extern "C" void kernel_launch(void* const* d_in, const int* in_sizes, int n_in,
                              void* d_out, int out_size, void* d_ws, size_t ws_size,
                              hipStream_t stream) {
  const float* x      = (const float*)d_in[0];
  const int*   ei     = (const int*)  d_in[1];
  const float* ea     = (const float*)d_in[2];
  const float* u      = (const float*)d_in[3];
  const float* Wp     = (const float*)d_in[4];
  const float* bp     = (const float*)d_in[5];
  const float* We1    = (const float*)d_in[6];
  const float* be1    = (const float*)d_in[7];
  const float* We2    = (const float*)d_in[8];
  const float* be2    = (const float*)d_in[9];
  const float* root   = (const float*)d_in[10];
  const float* conv_b = (const float*)d_in[11];
  const float* Wih    = (const float*)d_in[12];
  const float* bih    = (const float*)d_in[13];
  const float* Whh    = (const float*)d_in[14];
  const float* bhh    = (const float*)d_in[15];
  const float* Wd1    = (const float*)d_in[16];
  const float* bd1    = (const float*)d_in[17];
  const float* Wd2    = (const float*)d_in[18];
  const float* bd2    = (const float*)d_in[19];
  const float* Wd3    = (const float*)d_in[20];
  const float* bd3    = (const float*)d_in[21];
  float* out = (float*)d_out;

  // workspace layout (all 256B-aligned): ~34.3 MB total
  char* ws = (char*)d_ws;
  float*    h_buf   = (float*)   (ws + 0);
  float*    g_buf   = (float*)   (ws + 8388608);
  float*    msg_buf = (float*)   (ws + 16777216);
  float*    agg_buf = (float*)   (ws + 25165824);
  float*    counts  = (float*)   (ws + 33554432);
  _Float16* Bmsg    = (_Float16*)(ws + 33685504);
  _Float16* WB1p    = (_Float16*)(ws + 34217984);
  _Float16* WB2p    = (_Float16*)(ws + 34250752);
  _Float16* WD1p    = (_Float16*)(ws + 34275328);
  _Float16* WD2p    = (_Float16*)(ws + 34295808);

  pack_weights<<<1200, 256, 0, stream>>>(We2, be2, root, Whh, Wih, Wd1, Wd2,
                                         Bmsg, WB1p, WB2p, WD1p, WD2p);
  node_proj<<<(NN * HH) / 256, 256, 0, stream>>>(x, u, Wp, bp, h_buf);
  edge_gate<<<(EE * HH) / 256, 256, 0, stream>>>(ea, We1, be1, g_buf);
  hipMemsetAsync(counts, 0, NN * sizeof(float), stream);
  count_dst<<<EE / 256, 256, 0, stream>>>(ei, counts);

  for (int s = 0; s < 3; ++s) {
    hipMemsetAsync(agg_buf, 0, (size_t)NN * HH * sizeof(float), stream);
    msg_gemm<<<EE / 32, 32, 0, stream>>>(h_buf, g_buf, ei, Bmsg, msg_buf);
    scatter_msg<<<(EE * HH) / 256, 256, 0, stream>>>(msg_buf, ei, agg_buf);
    node_update<<<NN / 16, 32, 0, stream>>>(h_buf, agg_buf, counts, conv_b, bih, bhh, WB1p, WB2p);
  }

  edge_decode<<<EE / 16, 32, 0, stream>>>(h_buf, ea, ei, WD1p, WD2p, bd1, bd2, Wd3, bd3, out);
}
